// SwinTransformerBlock_58368605552720
// MI455X (gfx1250) — compile-verified, matrix operand order checked by reading, not run on hardware
//
#include <hip/hip_runtime.h>
#include <hip/hip_bf16.h>

// ---------------- types ----------------
typedef __bf16 bf16_t;
typedef bf16_t v16bf __attribute__((ext_vector_type(16)));
typedef bf16_t v8bf  __attribute__((ext_vector_type(8)));
typedef float  v8f   __attribute__((ext_vector_type(8)));
typedef float  v4f   __attribute__((ext_vector_type(4)));

#define B_  16
#define H_  56
#define C_  512
#define NH_ 16
#define DK_ 32
#define WS_ 7
#define SS_ 3
#define NTOK 50176        // B * 56*56  (also 1024 windows * 49)
#define HID_ 2048
static constexpr float SCALE_ = 0.17677669529663687f; // 32^-0.5

#define USE_ASYNC_LDS 1

#if defined(__has_builtin)
#if __has_builtin(__builtin_amdgcn_s_wait_asynccnt)
#define WAIT_ASYNCCNT(n) __builtin_amdgcn_s_wait_asynccnt(n)
#else
#define WAIT_ASYNCCNT(n) asm volatile("s_wait_asynccnt %0" :: "i"(n) : "memory")
#endif
#else
#define WAIT_ASYNCCNT(n) asm volatile("s_wait_asynccnt %0" :: "i"(n) : "memory")
#endif

__device__ __forceinline__ unsigned short f2bf(float f) {
    unsigned u = __builtin_bit_cast(unsigned, f);
    u += 0x7FFFu + ((u >> 16) & 1u);
    return (unsigned short)(u >> 16);
}

// async 16B copy global -> LDS (ASYNCcnt-tracked); lds: generic ptr, low 32 bits
// of the flat shared aperture address are the LDS byte offset.
__device__ __forceinline__ void async_ld_b128(void* lds, const void* g) {
    unsigned off = (unsigned)(size_t)lds;
    asm volatile("global_load_async_to_lds_b128 %0, %1, off"
                 :: "v"(off), "v"(g) : "memory");
}

// two 8-element runs at +0 and +16 (elements), per the 16-bit A/B VGPR layout
__device__ __forceinline__ v16bf load_frag(const unsigned short* p) {
    v8bf lo = *(const v8bf*)(const void*)(p);
    v8bf hi = *(const v8bf*)(const void*)(p + 16);
    return __builtin_shufflevector(lo, hi, 0,1,2,3,4,5,6,7,8,9,10,11,12,13,14,15);
}

// ---------------- LayerNorm kernels ----------------
__global__ __launch_bounds__(256) void ln1_kernel(const float* __restrict__ x,
                                                  const float* __restrict__ g,
                                                  const float* __restrict__ b,
                                                  unsigned short* __restrict__ out) {
    int wave = threadIdx.x >> 5, lane = threadIdx.x & 31;
    int m = blockIdx.x * 8 + wave;                 // window-ordered token
    int win = m / 49, nt = m - win * 49;
    int bb = win >> 6, wrem = win & 63;
    int wh = wrem >> 3, ww = wrem & 7;
    int i = nt / 7, j = nt - i * 7;
    int hh = wh * 7 + i + SS_; if (hh >= H_) hh -= H_;
    int wc = ww * 7 + j + SS_; if (wc >= H_) wc -= H_;
    const float* row = x + (((size_t)bb * H_ + hh) * H_ + wc) * C_;
    float v[16], s = 0.f, s2 = 0.f;
    #pragma unroll
    for (int e = 0; e < 16; ++e) { float t = row[lane * 16 + e]; v[e] = t; s += t; s2 += t * t; }
    #pragma unroll
    for (int off = 16; off; off >>= 1) { s += __shfl_xor(s, off); s2 += __shfl_xor(s2, off); }
    float mu = s * (1.f / C_);
    float rstd = rsqrtf(s2 * (1.f / C_) - mu * mu + 1e-5f);
    unsigned short* orow = out + (size_t)m * C_;
    #pragma unroll
    for (int e = 0; e < 16; ++e) {
        int c = lane * 16 + e;
        orow[c] = f2bf((v[e] - mu) * rstd * g[c] + b[c]);
    }
}

__global__ __launch_bounds__(256) void ln2_kernel(const float* __restrict__ x2,
                                                  const float* __restrict__ g,
                                                  const float* __restrict__ b,
                                                  unsigned short* __restrict__ out) {
    int wave = threadIdx.x >> 5, lane = threadIdx.x & 31;
    int m = blockIdx.x * 8 + wave;
    const float* row = x2 + (size_t)m * C_;
    float v[16], s = 0.f, s2 = 0.f;
    #pragma unroll
    for (int e = 0; e < 16; ++e) { float t = row[lane * 16 + e]; v[e] = t; s += t; s2 += t * t; }
    #pragma unroll
    for (int off = 16; off; off >>= 1) { s += __shfl_xor(s, off); s2 += __shfl_xor(s2, off); }
    float mu = s * (1.f / C_);
    float rstd = rsqrtf(s2 * (1.f / C_) - mu * mu + 1e-5f);
    unsigned short* orow = out + (size_t)m * C_;
    #pragma unroll
    for (int e = 0; e < 16; ++e) {
        int c = lane * 16 + e;
        orow[c] = f2bf((v[e] - mu) * rstd * g[c] + b[c]);
    }
}

// weight convert + transpose: src fp32 [K][N] -> dst bf16 [N][K]
__global__ void wconv_kernel(const float* __restrict__ src, unsigned short* __restrict__ dst,
                             int K, int N) {
    size_t idx = (size_t)blockIdx.x * 256 + threadIdx.x;
    if (idx < (size_t)K * N) {
        int n = (int)(idx % N);
        int k = (int)(idx / N);
        dst[(size_t)n * K + k] = f2bf(src[idx]);
    }
}

// bias+mask table, cls = (wh==7)*2 | (ww==7).
// Permuted inner layout: entry (m,n) stored at (m<<6) + (n&15)*4 + (n>>4), so a
// lane's 4 values (n = nt*16+lm, nt=0..3) are one contiguous float4.
__global__ void biastab_kernel(const float* __restrict__ rpb, float* __restrict__ tab) {
    int idx = blockIdx.x * 256 + threadIdx.x;      // < 4*16*64*64
    int n = idx & 63, m = (idx >> 6) & 63;
    int head = (idx >> 12) & 15, cls = idx >> 16;
    float v;
    if (m < 49 && n < 49) {
        int mi = m / 7, mj = m - mi * 7, ni = n / 7, nj = n - ni * 7;
        int bidx = (mi - ni + 6) * 13 + (mj - nj + 6);
        v = rpb[bidx * NH_ + head];
        int bh = (cls >> 1) & 1, bw = cls & 1;
        int rmh = bh ? (mi < 4 ? 1 : 2) : 0;
        int rmw = bw ? (mj < 4 ? 1 : 2) : 0;
        int rnh = bh ? (ni < 4 ? 1 : 2) : 0;
        int rnw = bw ? (nj < 4 ? 1 : 2) : 0;
        if (rmh * 3 + rmw != rnh * 3 + rnw) v -= 100.f;
    } else {
        v = (n >= 49) ? -1e30f : 0.f;
    }
    int pos = (idx & ~4095) | (m << 6) | ((n & 15) << 2) | (n >> 4);
    tab[pos] = v;
}

// ---------------- tiled WMMA GEMM (double-buffered, async-to-LDS) ----------------
// C[M][N] = A[M][K](bf16,row) x Bt[N][K](bf16,row) + bias ; epilogue by MODE
// MODE 0: store bf16 ; MODE 1: proj scatter+residual fp32 ; MODE 2: SiLU bf16 ;
// MODE 3: out_f[m*512+n] += v
template <int MODE>
__global__ __launch_bounds__(256) void gemm_kernel(const unsigned short* __restrict__ A,
                                                   const unsigned short* __restrict__ Bt,
                                                   const float* __restrict__ bias,
                                                   int K, int ldo,
                                                   unsigned short* __restrict__ out_bf,
                                                   float* __restrict__ out_f,
                                                   const float* __restrict__ resid) {
    __shared__ unsigned short As[2][128 * 40];
    __shared__ unsigned short Bs[2][128 * 40];
    const int t = threadIdx.x;
    const int wave = t >> 5, lane = t & 31;
    const int wrow = wave >> 2, wcol = wave & 3;       // 2 x 4 wave grid
    const int lm = lane & 15, kbase = (lane >> 4) * 8;
    const int tileM = blockIdx.y * 128, tileN = blockIdx.x * 128;

    v8f acc[4][2] = {};

#if USE_ASYNC_LDS
    // issue one tile stage (4 async b128 per thread: 2 for A, 2 for B)
    auto issue = [&](int k0, int buf) {
        #pragma unroll
        for (int c = t; c < 512; c += 256) {
            int row = c >> 2, col = (c & 3) * 8;
            async_ld_b128(&As[buf][row * 40 + col],
                          &A[(size_t)(tileM + row) * K + k0 + col]);
            async_ld_b128(&Bs[buf][row * 40 + col],
                          &Bt[(size_t)(tileN + row) * K + k0 + col]);
        }
    };
    issue(0, 0);
#endif

    for (int k0 = 0; k0 < K; k0 += 32) {
        const int buf = (k0 >> 5) & 1;
        const bool more = (k0 + 32) < K;
#if USE_ASYNC_LDS
        if (more) {
            issue(k0 + 32, buf ^ 1);
            if (k0 + 64 < K) {   // warm L2 two stages ahead
                __builtin_prefetch(&A[(size_t)(tileM + (t >> 1)) * K + k0 + 64], 0, 0);
                __builtin_prefetch(&Bt[(size_t)(tileN + (t >> 1)) * K + k0 + 64], 0, 0);
            }
            WAIT_ASYNCCNT(4);    // stage k0 complete (in-order), next stage in flight
        } else {
            WAIT_ASYNCCNT(0);
        }
        __syncthreads();
#else
        #pragma unroll
        for (int c = t; c < 512; c += 256) {
            int row = c >> 2, col = (c & 3) * 8;
            *(uint4*)(void*)&As[buf][row * 40 + col] =
                *(const uint4*)(const void*)&A[(size_t)(tileM + row) * K + k0 + col];
            *(uint4*)(void*)&Bs[buf][row * 40 + col] =
                *(const uint4*)(const void*)&Bt[(size_t)(tileN + row) * K + k0 + col];
        }
        __syncthreads();
#endif
        v16bf af[4], bfv[2];
        #pragma unroll
        for (int mt = 0; mt < 4; ++mt)
            af[mt] = load_frag(&As[buf][(wrow * 64 + mt * 16 + lm) * 40 + kbase]);
        #pragma unroll
        for (int nt = 0; nt < 2; ++nt)
            bfv[nt] = load_frag(&Bs[buf][(wcol * 32 + nt * 16 + lm) * 40 + kbase]);
        #pragma unroll
        for (int mt = 0; mt < 4; ++mt)
            #pragma unroll
            for (int nt = 0; nt < 2; ++nt)
                acc[mt][nt] = __builtin_amdgcn_wmma_f32_16x16x32_bf16(
                    false, af[mt], false, bfv[nt], (short)0, acc[mt][nt], false, false);
        __syncthreads();
    }

    // ---- epilogue ----
    #pragma unroll
    for (int mt = 0; mt < 4; ++mt) {
        int m0 = tileM + wrow * 64 + mt * 16 + (lane >> 4) * 8;
        int win = m0 / 49, ntok = m0 - win * 49;      // hoisted for MODE 1
        #pragma unroll
        for (int r = 0; r < 8; ++r) {
            int m = m0 + r;
            if (MODE == 1) {
                int bb = win >> 6, wrem = win & 63;
                int wh = wrem >> 3, ww = wrem & 7;
                int i = ntok / 7, j = ntok - i * 7;
                int hh = wh * 7 + i + SS_; if (hh >= H_) hh -= H_;
                int wc = ww * 7 + j + SS_; if (wc >= H_) wc -= H_;
                size_t base = (((size_t)bb * H_ + hh) * H_ + wc) * C_;
                #pragma unroll
                for (int nt = 0; nt < 2; ++nt) {
                    int n = tileN + wcol * 32 + nt * 16 + lm;
                    float v = acc[mt][nt][r] + bias[n];
                    out_f[base + n] = resid[base + n] + v;
                }
            } else {
                #pragma unroll
                for (int nt = 0; nt < 2; ++nt) {
                    int n = tileN + wcol * 32 + nt * 16 + lm;
                    float v = acc[mt][nt][r] + bias[n];
                    if (MODE == 0) {
                        out_bf[(size_t)m * ldo + n] = f2bf(v);
                    } else if (MODE == 2) {
                        float sv = v / (1.f + __expf(-v));
                        out_bf[(size_t)m * ldo + n] = f2bf(sv);
                    } else {
                        size_t idx = (size_t)m * C_ + n;
                        out_f[idx] = out_f[idx] + v;
                    }
                }
            }
            ntok++; if (ntok == 49) { ntok = 0; ++win; }
        }
    }
}

// ---------------- windowed attention (one wave per (window, head)) ----------------
__global__ __launch_bounds__(128) void attn_kernel(const unsigned short* __restrict__ qkv,
                                                   const float* __restrict__ tab,
                                                   unsigned short* __restrict__ attn_out) {
    __shared__ unsigned short sm[4 * (64 * 72 + 32 * 72)];
    const int wave = threadIdx.x >> 5, lane = threadIdx.x & 31;
    unsigned short* P  = &sm[wave * (64 * 72 + 32 * 72)];
    unsigned short* Vt = P + 64 * 72;
    const int gid = blockIdx.x * 4 + wave;     // 0 .. 16383
    const int win = gid >> 4, head = gid & 15;
    const int wrem = win & 63, wh = wrem >> 3, ww = wrem & 7;
    const int lm = lane & 15, kbase = (lane >> 4) * 8;
    const int cls = ((wh == 7) ? 2 : 0) | ((ww == 7) ? 1 : 0);
    const float* bt = tab + (((size_t)cls * NH_ + head) << 12) + (lm << 2);

    // ---- load Q and K fragments (pad rows >= 49 with zero) ----
    v16bf qf[4], kf[4];
    #pragma unroll
    for (int mt = 0; mt < 4; ++mt) {
        int m = mt * 16 + lm;
        v16bf z{};
        if (m < 49) {
            const unsigned short* p =
                qkv + ((size_t)(win * 49 + m) * 1536 + head * DK_ + kbase);
            z = load_frag(p);
        }
        qf[mt] = z;
        v16bf z2{};
        if (m < 49) {
            const unsigned short* p =
                qkv + ((size_t)(win * 49 + m) * 1536 + 512 + head * DK_ + kbase);
            z2 = load_frag(p);
        }
        kf[mt] = z2;
    }

    // ---- S = Q @ K^T (64x64, K=32) ----
    v8f s[4][4];
    #pragma unroll
    for (int mt = 0; mt < 4; ++mt)
        #pragma unroll
        for (int nt = 0; nt < 4; ++nt) {
            v8f c0 = {};
            s[mt][nt] = __builtin_amdgcn_wmma_f32_16x16x32_bf16(
                false, qf[mt], false, kf[nt], (short)0, c0, false, false);
        }

    // ---- scale + (bias+mask table, one float4 per row) + softmax -> P(bf16) ----
    #pragma unroll
    for (int mt = 0; mt < 4; ++mt) {
        #pragma unroll
        for (int r = 0; r < 8; ++r) {
            int m = mt * 16 + (lane >> 4) * 8 + r;
            v4f bm = *(const v4f*)(const void*)(bt + (m << 6));
            float v[4];
            float mx = -1e30f;
            #pragma unroll
            for (int nt = 0; nt < 4; ++nt) {
                float tv = fmaf(s[mt][nt][r], SCALE_, bm[nt]);
                v[nt] = tv;
                mx = fmaxf(mx, tv);
            }
            #pragma unroll
            for (int off = 8; off; off >>= 1) mx = fmaxf(mx, __shfl_xor(mx, off));
            float sum = 0.f;
            #pragma unroll
            for (int nt = 0; nt < 4; ++nt) { v[nt] = __expf(v[nt] - mx); sum += v[nt]; }
            #pragma unroll
            for (int off = 8; off; off >>= 1) sum += __shfl_xor(sum, off);
            float inv = 1.f / sum;
            #pragma unroll
            for (int nt = 0; nt < 4; ++nt)
                P[m * 72 + nt * 16 + lm] = f2bf(v[nt] * inv);
        }
    }

    // ---- V^T into LDS: Vt[d][n], n padded to 64 ----
    #pragma unroll
    for (int rep = 0; rep < 2; ++rep) {
        int n = lane + rep * 32;
        if (n < 49) {
            const unsigned short* vp =
                qkv + ((size_t)(win * 49 + n) * 1536 + 1024 + head * DK_);
            #pragma unroll
            for (int d = 0; d < 32; ++d) Vt[d * 72 + n] = vp[d];
        } else {
            #pragma unroll
            for (int d = 0; d < 32; ++d) Vt[d * 72 + n] = 0;
        }
    }
    __syncthreads();

    // ---- O = P @ V (64x32, K=64) ----
    v8f o[4][2] = {};
    #pragma unroll
    for (int kt = 0; kt < 2; ++kt) {
        v16bf pa[4], vb[2];
        #pragma unroll
        for (int mt = 0; mt < 4; ++mt)
            pa[mt] = load_frag(&P[(mt * 16 + lm) * 72 + kt * 32 + kbase]);
        #pragma unroll
        for (int dt = 0; dt < 2; ++dt)
            vb[dt] = load_frag(&Vt[(dt * 16 + lm) * 72 + kt * 32 + kbase]);
        #pragma unroll
        for (int mt = 0; mt < 4; ++mt)
            #pragma unroll
            for (int dt = 0; dt < 2; ++dt)
                o[mt][dt] = __builtin_amdgcn_wmma_f32_16x16x32_bf16(
                    false, pa[mt], false, vb[dt], (short)0, o[mt][dt], false, false);
    }

    // ---- store heads concatenated: attn_out[token][head*32+d] (bf16) ----
    #pragma unroll
    for (int mt = 0; mt < 4; ++mt)
        #pragma unroll
        for (int dt = 0; dt < 2; ++dt)
            #pragma unroll
            for (int r = 0; r < 8; ++r) {
                int m = mt * 16 + (lane >> 4) * 8 + r;
                if (m < 49) {
                    int d = dt * 16 + lm;
                    attn_out[(size_t)(win * 49 + m) * C_ + head * DK_ + d] =
                        f2bf(o[mt][dt][r]);
                }
            }
}

// ---------------- host launcher ----------------
extern "C" void kernel_launch(void* const* d_in, const int* in_sizes, int n_in,
                              void* d_out, int out_size, void* d_ws, size_t ws_size,
                              hipStream_t stream) {
    const float* x      = (const float*)d_in[0];
    const float* g1     = (const float*)d_in[1];
    const float* b1     = (const float*)d_in[2];
    const float* qkv_w  = (const float*)d_in[3];
    const float* qkv_b  = (const float*)d_in[4];
    const float* proj_w = (const float*)d_in[5];
    const float* proj_b = (const float*)d_in[6];
    const float* rpb    = (const float*)d_in[7];
    const float* g2     = (const float*)d_in[8];
    const float* b2     = (const float*)d_in[9];
    const float* w1     = (const float*)d_in[10];
    const float* b1m    = (const float*)d_in[11];
    const float* w2     = (const float*)d_in[12];
    const float* b2m    = (const float*)d_in[13];
    float* out = (float*)d_out;

    unsigned short* ws16 = (unsigned short*)d_ws;
    // workspace layout (ushort elements)
    unsigned short* qkvT  = ws16;                          // 1536*512
    unsigned short* projT = qkvT + 1536 * 512;             // 512*512
    unsigned short* w1T   = projT + 512 * 512;             // 2048*512
    unsigned short* w2T   = w1T + 2048 * 512;              // 512*2048
    float* biasTab        = (float*)(w2T + 512 * 2048);    // 4*16*64*64 fp32
    unsigned short* bufA  = (unsigned short*)(biasTab + 4 * 16 * 64 * 64); // NTOK*512
    unsigned short* bufR  = bufA + (size_t)NTOK * 512;     // region R
    unsigned short* bufQKV  = bufR;                        // NTOK*1536
    unsigned short* bufAttn = bufR + (size_t)NTOK * 1536;  // NTOK*512
    unsigned short* bufH    = bufR;                        // NTOK*2048 (reuses QKV+Attn)

    // 1. weights: convert + transpose to bf16 [N][K] ; bias+mask table
    wconv_kernel<<<(512 * 1536 + 255) / 256, 256, 0, stream>>>(qkv_w, qkvT, 512, 1536);
    wconv_kernel<<<(512 * 512 + 255) / 256, 256, 0, stream>>>(proj_w, projT, 512, 512);
    wconv_kernel<<<(512 * 2048 + 255) / 256, 256, 0, stream>>>(w1, w1T, 512, 2048);
    wconv_kernel<<<(2048 * 512 + 255) / 256, 256, 0, stream>>>(w2, w2T, 2048, 512);
    biastab_kernel<<<(4 * 16 * 64 * 64) / 256, 256, 0, stream>>>(rpb, biasTab);

    // 2. LN1 + shift + window partition -> bf16
    ln1_kernel<<<NTOK / 8, 256, 0, stream>>>(x, g1, b1, bufA);

    // 3. QKV GEMM
    gemm_kernel<0><<<dim3(1536 / 128, NTOK / 128), 256, 0, stream>>>(
        bufA, qkvT, qkv_b, 512, 1536, bufQKV, nullptr, nullptr);

    // 4. windowed attention
    attn_kernel<<<(1024 * NH_) / 4, 128, 0, stream>>>(bufQKV, biasTab, bufAttn);

    // 5. proj GEMM + window reverse + roll + residual -> x2 (fp32 in d_out)
    gemm_kernel<1><<<dim3(512 / 128, NTOK / 128), 256, 0, stream>>>(
        bufAttn, projT, proj_b, 512, 512, nullptr, out, x);

    // 6. LN2 on x2 -> bf16 (reuse bufA)
    ln2_kernel<<<NTOK / 8, 256, 0, stream>>>(out, g2, b2, bufA);

    // 7. MLP1 + SiLU -> bf16 hidden
    gemm_kernel<2><<<dim3(HID_ / 128, NTOK / 128), 256, 0, stream>>>(
        bufA, w1T, b1m, 512, HID_, bufH, nullptr, nullptr);

    // 8. MLP2 + residual accumulate into d_out
    gemm_kernel<3><<<dim3(512 / 128, NTOK / 128), 256, 0, stream>>>(
        bufH, w2T, b2m, HID_, 512, nullptr, out, nullptr);
}
